// SelfAttention_87514253623735
// MI455X (gfx1250) — matrix operand order repaired and checked by reading, not compiled, not run on hardware
//
#include <hip/hip_runtime.h>

#define BB 4
#define SS 4096
#define DDIM 1024
#define D3 3072

typedef __bf16 bf16_t;
typedef __attribute__((ext_vector_type(16))) __bf16 v16bf;
typedef __attribute__((ext_vector_type(8)))  __bf16 v8bf;
typedef __attribute__((ext_vector_type(8)))  float  v8f;

__device__ __forceinline__ v16bf cat8(v8bf lo, v8bf hi) {
  return __builtin_shufflevector(lo, hi, 0,1,2,3,4,5,6,7,8,9,10,11,12,13,14,15);
}

// ---- CDNA5 async global->LDS copy (ASYNCcnt path), with safe fallback ----
#if defined(__has_builtin)
#if __has_builtin(__builtin_amdgcn_global_load_async_to_lds_b128)
#define ASYNC_LDS_COPY 1
#endif
#endif

#ifdef ASYNC_LDS_COPY
// Real prototype (from hipcc diagnostic): param 1 is a pointer to a 128-bit
// int vector in the global (AS1, printed "__device__") address space.
typedef __attribute__((__vector_size__(4 * sizeof(int)))) int v4i_t;
typedef __attribute__((address_space(1))) v4i_t* g_b128_ptr;
typedef __attribute__((address_space(3))) v4i_t* l_b128_ptr;
#endif

__device__ __forceinline__ void async_copy_b128(const bf16_t* g, bf16_t* l) {
#ifdef ASYNC_LDS_COPY
  __builtin_amdgcn_global_load_async_to_lds_b128(
      (g_b128_ptr)(uintptr_t)g,
      (l_b128_ptr)(uint32_t)(uintptr_t)l,
      0, 0);
#else
  *(v8bf*)l = *(const v8bf*)g;
#endif
}

__device__ __forceinline__ void wait_async0() {
#ifdef ASYNC_LDS_COPY
#if __has_builtin(__builtin_amdgcn_s_wait_asynccnt)
  __builtin_amdgcn_s_wait_asynccnt(0);
#else
  asm volatile("s_wait_asynccnt 0x0" ::: "memory");
#endif
#endif
}

// ---------------------------------------------------------------- convert
__global__ void cvt_f32_bf16(const float* __restrict__ src,
                             bf16_t* __restrict__ dst, int n) {
  int i = blockIdx.x * blockDim.x + threadIdx.x;
  int stride = gridDim.x * blockDim.x;
  for (; i < n; i += stride) dst[i] = (bf16_t)src[i];
}

// ---------------------------------------------------------------- QKV GEMM
// C[m,n] = sum_k X[m,k] * W[n,k];  M = B*S = 16384, N = 3072, K = 1024.
// Block: 8 waves, each wave a 32x64 tile (same n0 across waves). The shared
// 64x32 B tile is double-buffered in LDS via async global->LDS loads.
// Q stored scaled by 1/sqrt(D)=1/32, K row-major, V transposed [b][d][s].
__global__ __launch_bounds__(256)
void qkv_gemm(const bf16_t* __restrict__ X, const bf16_t* __restrict__ W,
              bf16_t* __restrict__ Qs, bf16_t* __restrict__ Kb,
              bf16_t* __restrict__ Vt) {
  __shared__ bf16_t bbuf[2][64 * 32];

  const int tid   = threadIdx.x;
  const int lane  = tid & 31;
  const int wave  = tid >> 5;
  const int mt    = blockIdx.x * 8 + wave;      // 32-row tile index
  const int n0    = blockIdx.y * 64;
  const int ncol  = lane & 15;
  const int sel   = lane >> 4;
  const int rhalf = sel * 8;

  // B-staging assignment: thread copies one 16B chunk per k-step
  const int bn = tid >> 2;            // row 0..63 within n-tile
  const int bc = (tid & 3) * 8;       // element offset within 32-wide k slab
  const bf16_t* bsrc = W + (size_t)(n0 + bn) * DDIM + bc;
  const int bldsoff = bn * 32 + bc;

  v8f acc[8] = {};                    // [i_m(2)][j_n(4)]

  const bf16_t* arow0 = X + (size_t)(mt * 32 + ncol) * DDIM;
  const bf16_t* arow1 = arow0 + (size_t)16 * DDIM;

  async_copy_b128(bsrc, &bbuf[0][bldsoff]);     // prologue: k0 = 0
  int cur = 0;
  for (int k0 = 0; k0 < DDIM; k0 += 32) {
    wait_async0();
    __syncthreads();
    if (k0 + 32 < DDIM)
      async_copy_b128(bsrc + k0 + 32, &bbuf[cur ^ 1][bldsoff]);

    v16bf a0 = cat8(*(const v8bf*)(arow0 + k0 + sel * 16),
                    *(const v8bf*)(arow0 + k0 + sel * 16 + 8));
    v16bf a1 = cat8(*(const v8bf*)(arow1 + k0 + sel * 16),
                    *(const v8bf*)(arow1 + k0 + sel * 16 + 8));
#pragma unroll
    for (int j = 0; j < 4; ++j) {
      v16bf b = *(const v16bf*)&bbuf[cur][(j * 16 + ncol) * 32 + sel * 16];
      acc[j]     = __builtin_amdgcn_wmma_f32_16x16x32_bf16(
          false, a0, false, b, (short)0, acc[j], false, false);
      acc[4 + j] = __builtin_amdgcn_wmma_f32_16x16x32_bf16(
          false, a1, false, b, (short)0, acc[4 + j], false, false);
    }
    cur ^= 1;
  }

  const int mbase = mt * 32;
#pragma unroll
  for (int i = 0; i < 2; ++i) {
#pragma unroll
    for (int j = 0; j < 4; ++j) {
      int n = n0 + j * 16 + ncol;
#pragma unroll
      for (int r = 0; r < 8; ++r) {
        int m = mbase + i * 16 + rhalf + r;
        float c = acc[i * 4 + j][r];
        if (n < DDIM) {
          Qs[(size_t)m * DDIM + n] = (bf16_t)(c * 0.03125f);  // fold 1/sqrt(D)
        } else if (n < 2 * DDIM) {
          Kb[(size_t)m * DDIM + (n - DDIM)] = (bf16_t)c;
        } else {
          int d = n - 2 * DDIM;
          int b = m >> 12;            // / S
          int s = m & (SS - 1);
          Vt[((size_t)b * DDIM + d) * SS + s] = (bf16_t)c;
        }
      }
    }
  }
}

// ---------------------------------------------------------------- attention
// Workgroup: 8 waves, one 16-query tile; wave w owns d-slice [w*128, w*128+128).
// Flash-style online softmax over 32-key blocks. Logits reduced across waves
// through LDS (ds_add_f32); wave 0 computes softmax, publishes bf16 P + alpha/l.
__global__ __launch_bounds__(256)
void attn_kernel(const bf16_t* __restrict__ Qs, const bf16_t* __restrict__ Kb,
                 const bf16_t* __restrict__ Vt, bf16_t* __restrict__ O) {
  __shared__ float  lds_logits[16 * 32];
  __shared__ bf16_t lds_p[16 * 32];
  __shared__ float  lds_alpha[16];
  __shared__ float  lds_l[16];

  const int lane  = threadIdx.x & 31;
  const int wave  = threadIdx.x >> 5;
  const int b     = blockIdx.y;
  const int s0    = blockIdx.x << 4;
  const int ncol  = lane & 15;
  const int sel   = lane >> 4;
  const int rhalf = sel * 8;
  const int dd    = wave << 7;  // 128-wide d slice

  const bf16_t* qbase = Qs + ((size_t)b * SS + s0) * DDIM;
  const bf16_t* kbase = Kb + (size_t)b * SS * DDIM;
  const bf16_t* vbase = Vt + (size_t)b * DDIM * SS;

  v8f acc[8] = {};
  float mrow = -3.0e38f;  // live only in wave 0, lanes 0-15
  float lrow = 0.0f;

  for (int kk = 0; kk < SS; kk += 32) {
    // zero shared logits tile
    for (int i = threadIdx.x; i < 16 * 32; i += 256) lds_logits[i] = 0.0f;
    __syncthreads();

    // partial logits over our d-slice: S_part = Q[:,dd:dd+128] K[:,dd:dd+128]^T
    v8f pacc[2] = {};
#pragma unroll
    for (int t = 0; t < 4; ++t) {
      int kd = dd + t * 32;
      v16bf aq = cat8(*(const v8bf*)(qbase + (size_t)ncol * DDIM + kd + sel * 16),
                      *(const v8bf*)(qbase + (size_t)ncol * DDIM + kd + sel * 16 + 8));
#pragma unroll
      for (int j = 0; j < 2; ++j) {
        const bf16_t* kp =
            kbase + (size_t)(kk + j * 16 + ncol) * DDIM + kd + sel * 16;
        v16bf bk = *(const v16bf*)kp;
        pacc[j] = __builtin_amdgcn_wmma_f32_16x16x32_bf16(
            false, aq, false, bk, (short)0, pacc[j], false, false);
      }
    }
#pragma unroll
    for (int j = 0; j < 2; ++j)
#pragma unroll
      for (int r = 0; r < 8; ++r)
        atomicAdd(&lds_logits[(rhalf + r) * 32 + j * 16 + ncol], pacc[j][r]);
    __syncthreads();

    // wave 0: online softmax for the 16 rows (lane == row)
    if (wave == 0) {
      if (lane < 16) {
        float lv[32];
        float mx = mrow;
#pragma unroll
        for (int q = 0; q < 32; ++q) {
          float v = lds_logits[lane * 32 + q];
          lv[q] = v;
          mx = fmaxf(mx, v);
        }
        float al = __expf(mrow - mx);
        float s = 0.0f;
#pragma unroll
        for (int q = 0; q < 32; ++q) {
          float p = __expf(lv[q] - mx);
          s += p;
          lds_p[lane * 32 + q] = (bf16_t)p;
        }
        lrow = lrow * al + s;
        mrow = mx;
        lds_alpha[lane] = al;
        lds_l[lane] = lrow;
      }
    }
    __syncthreads();

    // rescale accumulators by alpha, then acc += P @ V_block
#pragma unroll
    for (int r = 0; r < 8; ++r) {
      float a = lds_alpha[rhalf + r];
#pragma unroll
      for (int jt = 0; jt < 8; ++jt) acc[jt][r] *= a;
    }
    v16bf pf = cat8(*(const v8bf*)(&lds_p[ncol * 32 + sel * 16]),
                    *(const v8bf*)(&lds_p[ncol * 32 + sel * 16 + 8]));
#pragma unroll
    for (int jt = 0; jt < 8; ++jt) {
      const bf16_t* vp =
          vbase + (size_t)(dd + jt * 16 + ncol) * SS + kk + sel * 16;
      v16bf bv = *(const v16bf*)vp;
      acc[jt] = __builtin_amdgcn_wmma_f32_16x16x32_bf16(
          false, pf, false, bv, (short)0, acc[jt], false, false);
    }
  }
  __syncthreads();

  // normalize and store bf16 attention output, row-major [b][s][d]
#pragma unroll
  for (int r = 0; r < 8; ++r) {
    float linv = 1.0f / lds_l[rhalf + r];
    bf16_t* orow = O + ((size_t)b * SS + s0 + rhalf + r) * DDIM;
#pragma unroll
    for (int jt = 0; jt < 8; ++jt)
      orow[dd + jt * 16 + ncol] = (bf16_t)(acc[jt][r] * linv);
  }
}

// ---------------------------------------------------------------- out GEMM
// Out[m,n] = sum_k A[m,k] * Wo[n,k];  M = 16384, N = K = 1024, fp32 output.
// Same structure as qkv_gemm: 32x64 wave tiles, async-staged shared B tile.
__global__ __launch_bounds__(256)
void out_gemm(const bf16_t* __restrict__ A, const bf16_t* __restrict__ W,
              float* __restrict__ Out) {
  __shared__ bf16_t bbuf[2][64 * 32];

  const int tid   = threadIdx.x;
  const int lane  = tid & 31;
  const int wave  = tid >> 5;
  const int mt    = blockIdx.x * 8 + wave;
  const int n0    = blockIdx.y * 64;
  const int ncol  = lane & 15;
  const int sel   = lane >> 4;
  const int rhalf = sel * 8;

  const int bn = tid >> 2;
  const int bc = (tid & 3) * 8;
  const bf16_t* bsrc = W + (size_t)(n0 + bn) * DDIM + bc;
  const int bldsoff = bn * 32 + bc;

  v8f acc[8] = {};

  const bf16_t* arow0 = A + (size_t)(mt * 32 + ncol) * DDIM;
  const bf16_t* arow1 = arow0 + (size_t)16 * DDIM;

  async_copy_b128(bsrc, &bbuf[0][bldsoff]);
  int cur = 0;
  for (int k0 = 0; k0 < DDIM; k0 += 32) {
    wait_async0();
    __syncthreads();
    if (k0 + 32 < DDIM)
      async_copy_b128(bsrc + k0 + 32, &bbuf[cur ^ 1][bldsoff]);

    v16bf a0 = cat8(*(const v8bf*)(arow0 + k0 + sel * 16),
                    *(const v8bf*)(arow0 + k0 + sel * 16 + 8));
    v16bf a1 = cat8(*(const v8bf*)(arow1 + k0 + sel * 16),
                    *(const v8bf*)(arow1 + k0 + sel * 16 + 8));
#pragma unroll
    for (int j = 0; j < 4; ++j) {
      v16bf b = *(const v16bf*)&bbuf[cur][(j * 16 + ncol) * 32 + sel * 16];
      acc[j]     = __builtin_amdgcn_wmma_f32_16x16x32_bf16(
          false, a0, false, b, (short)0, acc[j], false, false);
      acc[4 + j] = __builtin_amdgcn_wmma_f32_16x16x32_bf16(
          false, a1, false, b, (short)0, acc[4 + j], false, false);
    }
    cur ^= 1;
  }

  const int mbase = mt * 32;
#pragma unroll
  for (int i = 0; i < 2; ++i) {
#pragma unroll
    for (int j = 0; j < 4; ++j) {
      int n = n0 + j * 16 + ncol;
#pragma unroll
      for (int r = 0; r < 8; ++r)
        Out[(size_t)(mbase + i * 16 + rhalf + r) * DDIM + n] = acc[i * 4 + j][r];
    }
  }
}

// ---------------------------------------------------------------- launcher
extern "C" void kernel_launch(void* const* d_in, const int* in_sizes, int n_in,
                              void* d_out, int out_size, void* d_ws,
                              size_t ws_size, hipStream_t stream) {
  (void)in_sizes; (void)n_in; (void)out_size; (void)ws_size;

  const float* x    = (const float*)d_in[0];   // [B,S,D]
  const float* wqkv = (const float*)d_in[1];   // [3D,D]
  const float* wo   = (const float*)d_in[2];   // [D,D]
  float* out        = (float*)d_out;           // [B,S,D] fp32

  char* ws = (char*)d_ws;
  size_t off = 0;
  bf16_t* Xb    = (bf16_t*)(ws + off); off += (size_t)BB * SS * DDIM * 2;
  bf16_t* Wqkvb = (bf16_t*)(ws + off); off += (size_t)D3 * DDIM * 2;
  bf16_t* Wob   = (bf16_t*)(ws + off); off += (size_t)DDIM * DDIM * 2;
  bf16_t* Qb    = (bf16_t*)(ws + off); off += (size_t)BB * SS * DDIM * 2;
  bf16_t* Kbf   = (bf16_t*)(ws + off); off += (size_t)BB * SS * DDIM * 2;
  bf16_t* Vt    = (bf16_t*)(ws + off); off += (size_t)BB * SS * DDIM * 2;
  bf16_t* At    = (bf16_t*)(ws + off); off += (size_t)BB * SS * DDIM * 2;

  cvt_f32_bf16<<<4096, 256, 0, stream>>>(x, Xb, BB * SS * DDIM);
  cvt_f32_bf16<<<1024, 256, 0, stream>>>(wqkv, Wqkvb, D3 * DDIM);
  cvt_f32_bf16<<<512, 256, 0, stream>>>(wo, Wob, DDIM * DDIM);

  // M tiles: 16384/32 = 512, 8 waves/block -> 64 blocks; N: 3072/64 = 48
  qkv_gemm<<<dim3(64, 48), 256, 0, stream>>>(Xb, Wqkvb, Qb, Kbf, Vt);

  // one block per (16-query tile, batch)
  attn_kernel<<<dim3(SS / 16, BB), 256, 0, stream>>>(Qb, Kbf, Vt, At);

  // N: 1024/64 = 16
  out_gemm<<<dim3(64, 16), 256, 0, stream>>>(At, Wob, out);
}
